// NeuralDMD_Evaluation_23416161698147
// MI455X (gfx1250) — matrix operand order repaired and checked by reading, not compiled, not run on hardware
//
#include <hip/hip_runtime.h>

// ---------------------------------------------------------------------------
// Types for CDNA5 WMMA (wave32): D(f32 16x16) = A(bf16 16x32) x B(bf16 32x16) + C
// ---------------------------------------------------------------------------
typedef __attribute__((ext_vector_type(16))) __bf16 bf16x16;
typedef __attribute__((ext_vector_type(8)))  float  f32x8;
typedef __attribute__((ext_vector_type(4)))  unsigned int u32x4;
typedef __attribute__((ext_vector_type(4)))  int         i32x4;

union FragCast { u32x4 u[2]; bf16x16 v; };
union Pack8    { __bf16 b[8]; u32x4 u; };

#if defined(__gfx1250__) &&                                          \
    __has_builtin(__builtin_amdgcn_global_load_async_to_lds_b128) && \
    __has_builtin(__builtin_amdgcn_s_wait_asynccnt)
#define USE_ASYNC_LDS 1
#else
#define USE_ASYNC_LDS 0
#endif

__device__ __forceinline__ int lane_id() { return threadIdx.x & 31; }

// float -> bf16 (round-to-nearest-even), pure bit ops
__device__ __forceinline__ __bf16 f2bf(float x) {
  union { float f; unsigned int u; } a; a.f = x;
  unsigned int r = a.u + 0x7FFFu + ((a.u >> 16) & 1u);
  union { unsigned short s; __bf16 b; } o;
  o.s = (unsigned short)(r >> 16);
  return o.b;
}

// A fragment (16x32, MxK) from row-major bf16 [M][ld], tile at (row0, k0).
// Layout used here (same as frag_b): lane L -> m=L%16, half=L/16; elem e -> K = half*16 + e
__device__ __forceinline__ bf16x16 frag_a(const __bf16* p, int row0, int k0, int ld) {
  int l = lane_id();
  int m = l & 15, h = l >> 4;
  const __bf16* base = p + (size_t)(row0 + m) * ld + (k0 + h * 16);
  FragCast f;
  f.u[0] = *(const u32x4*)base;        // K = k0 + h*16 + [0..7]
  f.u[1] = *(const u32x4*)(base + 8);  // K = k0 + h*16 + [8..15]
  return f.v;
}

// B fragment (32x16, KxN) from B^T stored row-major [N][ld], tile at (n0, k0).
// Layout used here (same as frag_a): lane L -> n=L%16, half=L/16; elem e -> K = half*16 + e
__device__ __forceinline__ bf16x16 frag_b(const __bf16* bt, int n0, int k0, int ld) {
  int l = lane_id();
  int n = l & 15, h = l >> 4;
  const __bf16* base = bt + (size_t)(n0 + n) * ld + (k0 + h * 16);
  FragCast f;
  f.u[0] = *(const u32x4*)base;       // K = k0 + h*16 + [0..7]
  f.u[1] = *(const u32x4*)(base + 8); // K = k0 + h*16 + [8..15]
  return f.v;
}

// B fragment gathered (transpose-on-the-fly) from a row-major [K][ld] matrix.
// Used only for the 4 squarings of Atilde (negligible work).
__device__ __forceinline__ bf16x16 frag_b_tr(const __bf16* m_rm, int n0, int k0, int ld) {
  int l = lane_id();
  int n = l & 15, h = l >> 4;
  bf16x16 v;
#pragma unroll
  for (int e = 0; e < 16; ++e)
    v[e] = m_rm[(size_t)(k0 + h * 16 + e) * ld + (n0 + n)];
  return v;
}

__device__ __forceinline__ f32x8 wmma_bf16(bf16x16 a, bf16x16 b, f32x8 c) {
  return __builtin_amdgcn_wmma_f32_16x16x32_bf16(false, a, false, b, (short)0, c,
                                                 false, false);
}

// ---------------------------------------------------------------------------
// One-shot conversion / transpose kernels (weights are converted each launch)
// ---------------------------------------------------------------------------
__global__ void k_cvt(const float* __restrict__ in, __bf16* __restrict__ out, int n) {
  for (int i = threadIdx.x + blockIdx.x * 256; i < n; i += gridDim.x * 256)
    out[i] = f2bf(in[i]);
}

// out[C][R] = transpose(in[R][C]) with bf16 convert; R,C multiples of 32
__global__ __launch_bounds__(256) void k_cvt_t(const float* __restrict__ in,
                                               __bf16* __restrict__ out,
                                               int R, int C) {
  __shared__ float tile[32][33];
  int tx = threadIdx.x & 31;
  int ty = threadIdx.x >> 5;
  int tilesC = C / 32;
  int tr = (blockIdx.x / tilesC) * 32;
  int tc = (blockIdx.x % tilesC) * 32;
#pragma unroll
  for (int rr = ty; rr < 32; rr += 8)
    tile[rr][tx] = in[(size_t)(tr + rr) * C + (tc + tx)];
  __syncthreads();
#pragma unroll
  for (int rr = ty; rr < 32; rr += 8)
    out[(size_t)(tc + rr) * R + (tr + tx)] = f2bf(tile[tx][rr]);
}

// x0 [4][32768] f32 -> [16][32768] bf16, rows 4..15 zero-padded
__global__ void k_pad_x0(const float* __restrict__ x0, __bf16* __restrict__ out) {
  for (int i = threadIdx.x + blockIdx.x * 256; i < 16 * 32768; i += gridDim.x * 256) {
    int r = i >> 15;
    out[i] = (r < 4) ? f2bf(x0[i]) : f2bf(0.f);
  }
}

// ---------------------------------------------------------------------------
// Encoder layer 1: h1 = lrelu(x0 @ W0 + b0)  (M=16, K=32768, N=256)
// ---------------------------------------------------------------------------
__global__ __launch_bounds__(256) void k_enc1(const __bf16* __restrict__ x0b,
                                              const __bf16* __restrict__ W0t,
                                              const float* __restrict__ b0,
                                              __bf16* __restrict__ h1) {
  __shared__ float red[8][16][16];
  const int wid = threadIdx.x >> 5;
  const int l = lane_id(), n = l & 15, h = l >> 4;
  const int nt = blockIdx.x;
  f32x8 c = {};
  for (int kc = wid; kc < 1024; kc += 8)
    c = wmma_bf16(frag_a(x0b, 0, kc * 32, 32768),
                  frag_b(W0t, nt * 16, kc * 32, 32768), c);
#pragma unroll
  for (int v = 0; v < 8; ++v) red[wid][v + 8 * h][n] = c[v];
  __syncthreads();
  int m  = threadIdx.x >> 4;
  int nn = threadIdx.x & 15;
  float s = 0.f;
#pragma unroll
  for (int w = 0; w < 8; ++w) s += red[w][m][nn];
  s += b0[nt * 16 + nn];
  s = (s >= 0.f) ? s : 0.01f * s;
  h1[(size_t)m * 256 + nt * 16 + nn] = f2bf(s);
}

// ---------------------------------------------------------------------------
// Encoder layers 2..4 -> z0 (padded [16][256] bf16; last layer: no activation)
// ---------------------------------------------------------------------------
__global__ __launch_bounds__(256) void k_enc_rest(
    const __bf16* __restrict__ h1,
    const __bf16* __restrict__ W1t, const float* __restrict__ b1,
    const __bf16* __restrict__ W2t, const float* __restrict__ b2,
    const __bf16* __restrict__ W3t, const float* __restrict__ b3,
    __bf16* __restrict__ z0b) {
  __shared__ __attribute__((aligned(16))) __bf16 X[2][16 * 256];
  const int wid = threadIdx.x >> 5;
  const int l = lane_id(), n = l & 15, h = l >> 4;
  for (int i = threadIdx.x; i < (16 * 256) / 8; i += 256)
    ((u32x4*)X[0])[i] = ((const u32x4*)h1)[i];
  __syncthreads();
  const __bf16* Wt[3] = {W1t, W2t, W3t};
  const float*  bs[3] = {b1, b2, b3};
#pragma unroll
  for (int lay = 0; lay < 3; ++lay) {
    bf16x16 afr[8];
#pragma unroll
    for (int kc = 0; kc < 8; ++kc) afr[kc] = frag_a(X[lay & 1], 0, kc * 32, 256);
#pragma unroll
    for (int t = 0; t < 2; ++t) {
      int nt = wid * 2 + t;
      f32x8 c = {};
#pragma unroll
      for (int kc = 0; kc < 8; ++kc)
        c = wmma_bf16(afr[kc], frag_b(Wt[lay], nt * 16, kc * 32, 256), c);
      float bb = bs[lay][nt * 16 + n];
#pragma unroll
      for (int v = 0; v < 8; ++v) {
        float x = c[v] + bb;
        if (lay < 2) {
          x = (x >= 0.f) ? x : 0.01f * x;
          X[(lay + 1) & 1][(size_t)(v + 8 * h) * 256 + nt * 16 + n] = f2bf(x);
        } else {
          z0b[(size_t)(v + 8 * h) * 256 + nt * 16 + n] = f2bf(x);
        }
      }
    }
    __syncthreads();
  }
}

// ---------------------------------------------------------------------------
// Chunked DMD scan (single workgroup, 8 waves):
//   w0 = U^T z0^T; 15 small steps; M <- A^16 (4 squarings); 31 chunk steps.
// Emits V^T rows: Vt[t*4+b][r] = v_t[r][b].
// ---------------------------------------------------------------------------
__global__ __launch_bounds__(256) void k_scan(const __bf16* __restrict__ Ab,
                                              const __bf16* __restrict__ Ut,
                                              const __bf16* __restrict__ z0b,
                                              __bf16* __restrict__ Vt) {
  __shared__ __attribute__((aligned(16))) __bf16 Mr[128 * 128];   // 32 KB
  __shared__ __attribute__((aligned(16))) __bf16 VcT[80 * 128];   // 20 KB (padded)
  const int wid = threadIdx.x >> 5;
  const int l = lane_id(), n = l & 15, h = l >> 4;
  const int row0 = wid * 16;
  const int col = row0 + 8 * h;

  for (int i = threadIdx.x; i < (128 * 128) / 8; i += 256)
    ((u32x4*)Mr)[i] = ((const u32x4*)Ab)[i];

  // ---- w0 ----
  {
    f32x8 c = {};
#pragma unroll
    for (int kc = 0; kc < 8; ++kc)
      c = wmma_bf16(frag_a(Ut, row0, kc * 32, 256), frag_b(z0b, 0, kc * 32, 256), c);
    __syncthreads();
    if (n < 4) {
      Pack8 pk;
#pragma unroll
      for (int v = 0; v < 8; ++v) pk.b[v] = f2bf(c[v]);
      *(u32x4*)(VcT + (size_t)n * 128 + col) = pk.u;
      *(u32x4*)(Vt  + (size_t)n * 128 + col) = pk.u;
    }
    __syncthreads();
  }

  // ---- 15 sequential small steps ----
  {
    bf16x16 afr[4];
#pragma unroll
    for (int kc = 0; kc < 4; ++kc) afr[kc] = frag_a(Ab, row0, kc * 32, 128);
    for (int t = 1; t < 16; ++t) {
      f32x8 c = {};
#pragma unroll
      for (int kc = 0; kc < 4; ++kc)
        c = wmma_bf16(afr[kc], frag_b(VcT, (t - 1) * 4, kc * 32, 128), c);
      __syncthreads();
      if (n < 4) {
        Pack8 pk;
#pragma unroll
        for (int v = 0; v < 8; ++v) pk.b[v] = f2bf(c[v]);
        *(u32x4*)(VcT + (size_t)(t * 4 + n) * 128 + col) = pk.u;
        *(u32x4*)(Vt  + (size_t)(t * 4 + n) * 128 + col) = pk.u;
      }
      __syncthreads();
    }
  }

  // ---- M <- A^16 via 4 squarings ----
  for (int s = 0; s < 4; ++s) {
    bf16x16 aq[4];
#pragma unroll
    for (int kc = 0; kc < 4; ++kc) aq[kc] = frag_a(Mr, row0, kc * 32, 128);
    f32x8 cs[8];
    f32x8 zero = {};
#pragma unroll
    for (int nt = 0; nt < 8; ++nt) {
      cs[nt] = zero;
#pragma unroll
      for (int kc = 0; kc < 4; ++kc)
        cs[nt] = wmma_bf16(aq[kc], frag_b_tr(Mr, nt * 16, kc * 32, 128), cs[nt]);
    }
    __syncthreads();
#pragma unroll
    for (int nt = 0; nt < 8; ++nt)
#pragma unroll
      for (int v = 0; v < 8; ++v)
        Mr[(size_t)(row0 + v + 8 * h) * 128 + nt * 16 + n] = f2bf(cs[nt][v]);
    __syncthreads();
  }

  // ---- 31 chunk steps: advance 16 timesteps (64 V^T rows) at once ----
  {
    bf16x16 am[4];
#pragma unroll
    for (int kc = 0; kc < 4; ++kc) am[kc] = frag_a(Mr, row0, kc * 32, 128);
    for (int j = 1; j < 32; ++j) {
      f32x8 cc[4];
      f32x8 zero = {};
#pragma unroll
      for (int nt = 0; nt < 4; ++nt) {
        cc[nt] = zero;
#pragma unroll
        for (int kc = 0; kc < 4; ++kc)
          cc[nt] = wmma_bf16(am[kc], frag_b(VcT, nt * 16, kc * 32, 128), cc[nt]);
      }
      __syncthreads();
#pragma unroll
      for (int nt = 0; nt < 4; ++nt) {
        Pack8 pk;
#pragma unroll
        for (int v = 0; v < 8; ++v) pk.b[v] = f2bf(cc[nt][v]);
        int r = nt * 16 + n;
        *(u32x4*)(VcT + (size_t)r * 128 + col)            = pk.u;
        *(u32x4*)(Vt  + (size_t)(j * 64 + r) * 128 + col) = pk.u;
      }
      __syncthreads();
    }
  }
}

// ---------------------------------------------------------------------------
// Z[2048][256] = V^T @ U^T   (A = Vt rows, B^T = U itself)
// ---------------------------------------------------------------------------
__global__ __launch_bounds__(256) void k_zgemm(const __bf16* __restrict__ Vt,
                                               const __bf16* __restrict__ Ub,
                                               __bf16* __restrict__ Z) {
  const int wid = threadIdx.x >> 5;
  const int l = lane_id(), n = l & 15, h = l >> 4;
  const int row0 = blockIdx.x * 16;
  bf16x16 afr[4];
#pragma unroll
  for (int kc = 0; kc < 4; ++kc) afr[kc] = frag_a(Vt, row0, kc * 32, 128);
#pragma unroll
  for (int t = 0; t < 2; ++t) {
    int nt = wid * 2 + t;
    f32x8 c = {};
#pragma unroll
    for (int kc = 0; kc < 4; ++kc)
      c = wmma_bf16(afr[kc], frag_b(Ub, nt * 16, kc * 32, 128), c);
#pragma unroll
    for (int v = 0; v < 8; ++v)
      Z[(size_t)(row0 + v + 8 * h) * 256 + nt * 16 + n] = f2bf(c[v]);
  }
}

// First 4 Z rows are z0 itself (t=0 block), overwrite after k_zgemm.
__global__ void k_zfix(const __bf16* __restrict__ z0b, __bf16* __restrict__ Z) {
  int i = threadIdx.x + blockIdx.x * blockDim.x;
  if (i < 4 * 256) Z[i] = z0b[i];
}

// ---------------------------------------------------------------------------
// Fused decoder. Grid (32 row-blocks of 64 rows, 8 n-groups of 4096 cols).
// Phase 1: 3 hidden layers in a single 32 KB LDS buffer (read/barrier/write).
// Phase 2: A-fragments hoisted to registers; the same 32 KB LDS is reused as a
//          double-buffered weight stage filled with GLOBAL_LOAD_ASYNC_TO_LDS
//          (ASYNCcnt pipeline), shared by all 4 M-tiles -> 4x less L2 traffic.
// ---------------------------------------------------------------------------
__global__ __launch_bounds__(256) void k_decoder(
    const __bf16* __restrict__ Z,
    const __bf16* __restrict__ W1t, const float* __restrict__ b1,
    const __bf16* __restrict__ W2t, const float* __restrict__ b2,
    const __bf16* __restrict__ W3t, const float* __restrict__ b3,
    const __bf16* __restrict__ W4t, const float* __restrict__ b4,
    float* __restrict__ out) {
  __shared__ __attribute__((aligned(16))) __bf16 smem[16384];  // 32 KB, two roles
  const int wid = threadIdx.x >> 5;
  const int l = lane_id(), n = l & 15, h = l >> 4;
  const int row0 = blockIdx.x * 64;
  const int mt = wid >> 1;          // wave's M-tile (0..3)
  const int nsub = wid & 1;         // wave's N sub-slot

  // ---- phase 1: hidden layers for this block's 64 rows ----
  __bf16* X = smem;                 // [64][256] bf16
  {
    const __bf16* Wt[3] = {W1t, W2t, W3t};
    const float*  bs[3] = {b1, b2, b3};
    const int nbase = nsub * 8;     // 8 n-tiles per wave
#pragma unroll
    for (int lay = 0; lay < 3; ++lay) {
      const __bf16* src = (lay == 0) ? (Z + (size_t)row0 * 256) : X;
      bf16x16 afr[8];
#pragma unroll
      for (int kc = 0; kc < 8; ++kc) afr[kc] = frag_a(src, mt * 16, kc * 32, 256);
      f32x8 cs[8];
#pragma unroll
      for (int t = 0; t < 8; ++t) {
        f32x8 c = {};
#pragma unroll
        for (int kc = 0; kc < 8; ++kc)
          c = wmma_bf16(afr[kc], frag_b(Wt[lay], (nbase + t) * 16, kc * 32, 256), c);
        cs[t] = c;
      }
      __syncthreads();              // all reads of X done
#pragma unroll
      for (int t = 0; t < 8; ++t) {
        int nt = nbase + t;
        float bb = bs[lay][nt * 16 + n];
#pragma unroll
        for (int v = 0; v < 8; ++v) {
          float x = cs[t][v] + bb;
          x = (x >= 0.f) ? x : 0.01f * x;
          X[(size_t)(mt * 16 + v + 8 * h) * 256 + nt * 16 + n] = f2bf(x);
        }
      }
      __syncthreads();              // X updated for next layer
    }
  }

  // ---- hoist final-layer A fragments, then retire X ----
  bf16x16 afr[8];
#pragma unroll
  for (int kc = 0; kc < 8; ++kc) afr[kc] = frag_a(X, mt * 16, kc * 32, 256);
  __syncthreads();                  // smem now reusable as weight stage

  // ---- phase 2: out = h3 @ W4 + b4 over this block's 4096 columns ----
  __bf16* Wb = smem;                // [2][32][256] : two 16 KB slabs
  const int ntile0 = blockIdx.y * 256;   // 256 n-tiles per group, 2 per iter

  auto stage = [&](int buf, int iter) {
    const __bf16* g = W4t + (size_t)(ntile0 + 2 * iter) * 16 * 256;
    __bf16* d = Wb + buf * (32 * 256);
#if USE_ASYNC_LDS
#pragma unroll
    for (int q = 0; q < 4; ++q) {
      int j = threadIdx.x + q * 256;
      __builtin_amdgcn_global_load_async_to_lds_b128(
          (i32x4*)(g + j * 8), (i32x4*)(d + j * 8), 0, 0);
    }
#else
#pragma unroll
    for (int q = 0; q < 4; ++q) {
      int j = threadIdx.x + q * 256;
      *(u32x4*)(d + j * 8) = *(const u32x4*)(g + j * 8);
    }
#endif
  };

  stage(0, 0);
  for (int i = 0; i < 128; ++i) {
    if (i + 1 < 128) {
      stage((i + 1) & 1, i + 1);
#if USE_ASYNC_LDS
      __builtin_amdgcn_s_wait_asynccnt(4);  // slab i landed; slab i+1 in flight
#endif
    } else {
#if USE_ASYNC_LDS
      __builtin_amdgcn_s_wait_asynccnt(0);
#endif
    }
    __syncthreads();                // publish slab i to all waves
    const __bf16* wslab = Wb + (i & 1) * (32 * 256) + nsub * (16 * 256);
    f32x8 c = {};
#pragma unroll
    for (int kc = 0; kc < 8; ++kc)
      c = wmma_bf16(afr[kc], frag_b(wslab, 0, kc * 32, 256), c);
    int ncol = (ntile0 + 2 * i + nsub) * 16 + n;
    float bb = b4[ncol];
    float* o = out + (size_t)(row0 + mt * 16 + 8 * h) * 32768 + ncol;
#pragma unroll
    for (int v = 0; v < 8; ++v)
      o[(size_t)v * 32768] = c[v] + bb;
    __syncthreads();                // done reading slab i before it is refilled
  }
}

// ---------------------------------------------------------------------------
// Host launcher
// ---------------------------------------------------------------------------
extern "C" void kernel_launch(void* const* d_in, const int* in_sizes, int n_in,
                              void* d_out, int out_size, void* d_ws, size_t ws_size,
                              hipStream_t stream) {
  (void)in_sizes; (void)n_in; (void)out_size; (void)ws_size;
  const float* x0  = (const float*)d_in[0];
  // d_in[1] = idx (consecutive 0..T-1, unused — same assumption as reference)
  const float* Atl = (const float*)d_in[2];
  const float* U   = (const float*)d_in[3];
  const float* eW[4] = {(const float*)d_in[4],  (const float*)d_in[5],
                        (const float*)d_in[6],  (const float*)d_in[7]};
  const float* eB[4] = {(const float*)d_in[8],  (const float*)d_in[9],
                        (const float*)d_in[10], (const float*)d_in[11]};
  const float* dW[4] = {(const float*)d_in[12], (const float*)d_in[13],
                        (const float*)d_in[14], (const float*)d_in[15]};
  const float* dB[4] = {(const float*)d_in[16], (const float*)d_in[17],
                        (const float*)d_in[18], (const float*)d_in[19]};
  float* out = (float*)d_out;

  char* base = (char*)d_ws;
  size_t off = 0;
  auto take = [&](size_t elems) {
    __bf16* p = (__bf16*)(base + off);
    off += ((elems * 2 + 255) & ~(size_t)255);
    return p;
  };
  __bf16* x0b  = take((size_t)16 * 32768);    // padded x0, bf16
  __bf16* eW0t = take((size_t)256 * 32768);   // enc W0^T
  __bf16* eW1t = take(256 * 256);
  __bf16* eW2t = take(256 * 256);
  __bf16* eW3t = take(256 * 256);
  __bf16* dW0t = take(256 * 256);
  __bf16* dW1t = take(256 * 256);
  __bf16* dW2t = take(256 * 256);
  __bf16* dW3t = take((size_t)32768 * 256);   // dec W3^T
  __bf16* Ub   = take(256 * 128);             // U as-is  (B^T for Z GEMM)
  __bf16* Utp  = take(256 * 128);             // U^T      (A for w0 GEMM)
  __bf16* Ab   = take(128 * 128);             // Atilde bf16
  __bf16* h1   = take(16 * 256);
  __bf16* z0b  = take(16 * 256);
  __bf16* Vtb  = take((size_t)2048 * 128);    // V^T rows, bf16
  __bf16* Zb   = take((size_t)2048 * 256);    // latent trajectory, bf16

  // conversions / transposes
  k_pad_x0<<<512, 256, 0, stream>>>(x0, x0b);
  k_cvt<<<64, 256, 0, stream>>>(Atl, Ab, 128 * 128);
  k_cvt<<<128, 256, 0, stream>>>(U, Ub, 256 * 128);
  k_cvt_t<<<(256 / 32) * (128 / 32), 256, 0, stream>>>(U, Utp, 256, 128);
  k_cvt_t<<<(32768 / 32) * (256 / 32), 256, 0, stream>>>(eW[0], eW0t, 32768, 256);
  k_cvt_t<<<64, 256, 0, stream>>>(eW[1], eW1t, 256, 256);
  k_cvt_t<<<64, 256, 0, stream>>>(eW[2], eW2t, 256, 256);
  k_cvt_t<<<64, 256, 0, stream>>>(eW[3], eW3t, 256, 256);
  k_cvt_t<<<64, 256, 0, stream>>>(dW[0], dW0t, 256, 256);
  k_cvt_t<<<64, 256, 0, stream>>>(dW[1], dW1t, 256, 256);
  k_cvt_t<<<64, 256, 0, stream>>>(dW[2], dW2t, 256, 256);
  k_cvt_t<<<(256 / 32) * (32768 / 32), 256, 0, stream>>>(dW[3], dW3t, 256, 32768);

  // pipeline
  k_enc1<<<16, 256, 0, stream>>>(x0b, eW0t, eB[0], h1);
  k_enc_rest<<<1, 256, 0, stream>>>(h1, eW1t, eB[1], eW2t, eB[2], eW3t, eB[3], z0b);
  k_scan<<<1, 256, 0, stream>>>(Ab, Utp, z0b, Vtb);
  k_zgemm<<<128, 256, 0, stream>>>(Vtb, Ub, Zb);
  k_zfix<<<4, 256, 0, stream>>>(z0b, Zb);
  k_decoder<<<dim3(32, 8), 256, 0, stream>>>(Zb, dW0t, dB[0], dW1t, dB[1],
                                             dW2t, dB[2], dW3t, dB[3], out);
}
